// BitfieldLinear_1073741824329
// MI455X (gfx1250) — compile-verified, matrix operand order checked by reading, not run on hardware
//
#include <hip/hip_runtime.h>

// BitfieldLinear for MI455X (gfx1250, wave32, WMMA).
// y[m,o] = sum_i x[m,i] * basis[codes[o,i]] * scales[o] + bias[o]
// M=8192 (4*2048), N=OUT_F=4096, K=IN_F=4096.
// Workspace layout: [0, 32MiB) = W decoded to f16 ; [32MiB, 96MiB) = x in f16.

typedef _Float16 v16h __attribute__((ext_vector_type(16)));
typedef _Float16 v8h  __attribute__((ext_vector_type(8)));
typedef _Float16 v4h  __attribute__((ext_vector_type(4)));
typedef float    v8f  __attribute__((ext_vector_type(8)));
typedef int      v4i  __attribute__((ext_vector_type(4)));

#define IN_F   4096
#define OUT_F  4096
#define NROWS  8192

#define AS1 __attribute__((address_space(1)))
#define AS3 __attribute__((address_space(3)))

// ---------------------------------------------------------------- decode W
__global__ __launch_bounds__(256) void decode_w_kernel(
    const int* __restrict__ codes, const float* __restrict__ basis,
    const float* __restrict__ scales, _Float16* __restrict__ Wh)
{
  __shared__ float sb[256];
  sb[threadIdx.x] = basis[threadIdx.x];
  __syncthreads();
  size_t t = (size_t)blockIdx.x * 256 + threadIdx.x;     // one int4 of codes
  int4 c = ((const int4*)codes)[t];
  int o = (int)((t * 4) >> 12);                          // row length 4096
  float s = scales[o];
  v4h h;
  h.x = (_Float16)(sb[c.x] * s);
  h.y = (_Float16)(sb[c.y] * s);
  h.z = (_Float16)(sb[c.z] * s);
  h.w = (_Float16)(sb[c.w] * s);
  ((v4h*)Wh)[t] = h;
}

// ---------------------------------------------------------------- x -> f16
__global__ __launch_bounds__(256) void convert_x_kernel(
    const float* __restrict__ x, _Float16* __restrict__ xh)
{
  size_t t = (size_t)blockIdx.x * 256 + threadIdx.x;     // 8 floats per thread
  float4 a = ((const float4*)x)[2 * t];
  float4 b = ((const float4*)x)[2 * t + 1];
  v8h h;
  h[0] = (_Float16)a.x; h[1] = (_Float16)a.y; h[2] = (_Float16)a.z; h[3] = (_Float16)a.w;
  h[4] = (_Float16)b.x; h[5] = (_Float16)b.y; h[6] = (_Float16)b.z; h[7] = (_Float16)b.w;
  ((v8h*)xh)[t] = h;
}

// ---------------------------------------------------------------- GEMM
constexpr int BM = 128, BN = 128, BK = 64;
constexpr int LDT = BK + 8;      // halves per LDS row (+16B pad vs bank conflicts)
constexpr int KT  = IN_F / BK;   // 64 k-steps

#if __has_builtin(__builtin_amdgcn_global_load_async_to_lds_b128)
#define USE_ASYNC_LDS 1
#else
#define USE_ASYNC_LDS 0
#endif

struct Frag2 { v8h lo, hi; };

// Gather one 16x32 f16 WMMA operand fragment from a row-major LDS tile.
// Per ISA layout: lanes 0-15 hold K 0..7 / 16..23, lanes 16-31 hold K 8..15 / 24..31.
__device__ __forceinline__ v16h load_frag(const _Float16* lds, int row, int kk, int lane)
{
  const int hi = (lane >> 4) << 3;            // 0 or 8
  const _Float16* p = lds + row * LDT + kk + hi;
  Frag2 f;
  f.lo = *(const v8h*)(p);                    // ds_load_b128
  f.hi = *(const v8h*)(p + 16);               // ds_load_b128
  return __builtin_bit_cast(v16h, f);
}

// Stage ROWS x BK halves (row-major, src leading dim IN_F) into LDS.
// 256 threads: 2 threads per row, 64B each = 4 x b128 per thread.
__device__ __forceinline__ void stage_tile(_Float16* dst, const _Float16* src, int tid)
{
  const int row  = tid >> 1;
  const int half = tid & 1;
  const _Float16* g = src + (size_t)row * IN_F + half * 32;
  _Float16*       l = dst + row * LDT + half * 32;
#if USE_ASYNC_LDS
  #pragma unroll
  for (int j = 0; j < 4; ++j) {
    __builtin_amdgcn_global_load_async_to_lds_b128(
        (AS1 v4i*)(uintptr_t)(g + j * 8),
        (AS3 v4i*)(uint32_t)(uintptr_t)(l + j * 8),
        0, 0);
  }
#else
  #pragma unroll
  for (int j = 0; j < 4; ++j)
    *(v8h*)(l + j * 8) = *(const v8h*)(g + j * 8);
#endif
}

__device__ __forceinline__ void wait_stage()
{
#if USE_ASYNC_LDS
  asm volatile("s_wait_asynccnt 0x0" ::: "memory");
#endif
}

__global__ __launch_bounds__(256) void gemm_kernel(
    const _Float16* __restrict__ Xh, const _Float16* __restrict__ Wh,
    const float* __restrict__ bias, float* __restrict__ out)
{
  __shared__ _Float16 lA[2][BM * LDT];
  __shared__ _Float16 lB[2][BN * LDT];

  const int tid  = threadIdx.x;
  const int lane = tid & 31;
  const int wave = tid >> 5;          // 8 waves
  const int wm   = wave & 3;          // 4 waves in M -> 32 rows each
  const int wn   = wave >> 2;         // 2 waves in N -> 64 cols each
  const int m_blk = blockIdx.y * BM;
  const int n_blk = blockIdx.x * BN;

  v8f acc[2][4];
  #pragma unroll
  for (int mt = 0; mt < 2; ++mt)
    #pragma unroll
    for (int nt = 0; nt < 4; ++nt)
      #pragma unroll
      for (int v = 0; v < 8; ++v) acc[mt][nt][v] = 0.0f;

  const _Float16* gA = Xh + (size_t)m_blk * IN_F;   // A tile origin (x)
  const _Float16* gB = Wh + (size_t)n_blk * IN_F;   // B tile origin (W rows = B cols)

  // prologue: stage k-step 0 into buffer 0
  stage_tile(lA[0], gA, tid);
  stage_tile(lB[0], gB, tid);

  for (int kt = 0; kt < KT; ++kt) {
    wait_stage();
    __syncthreads();            // buffer kt&1 ready for all waves

    if (kt + 1 < KT) {          // overlap: stage next k-step while computing
      const int nb = (kt + 1) & 1;
      stage_tile(lA[nb], gA + (kt + 1) * BK, tid);
      stage_tile(lB[nb], gB + (kt + 1) * BK, tid);
    }

    const _Float16* A = lA[kt & 1];
    const _Float16* B = lB[kt & 1];

    #pragma unroll
    for (int kk = 0; kk < BK; kk += 32) {
      v16h af[2], bf[4];
      #pragma unroll
      for (int mt = 0; mt < 2; ++mt)
        af[mt] = load_frag(A, wm * 32 + mt * 16 + (lane & 15), kk, lane);
      #pragma unroll
      for (int nt = 0; nt < 4; ++nt)
        bf[nt] = load_frag(B, wn * 64 + nt * 16 + (lane & 15), kk, lane);
      #pragma unroll
      for (int mt = 0; mt < 2; ++mt)
        #pragma unroll
        for (int nt = 0; nt < 4; ++nt)
          acc[mt][nt] = __builtin_amdgcn_wmma_f32_16x16x32_f16(
              false, af[mt], false, bf[nt], (short)0, acc[mt][nt], false, false);
    }
  }

  // epilogue: C/D layout -> lane L, VGPR v holds (M = v + (L>=16 ? 8:0), N = L%16)
  #pragma unroll
  for (int mt = 0; mt < 2; ++mt) {
    const int row0 = m_blk + wm * 32 + mt * 16 + ((lane >> 4) << 3);
    #pragma unroll
    for (int nt = 0; nt < 4; ++nt) {
      const int col = n_blk + wn * 64 + nt * 16 + (lane & 15);
      const float bv = bias[col];
      float* o = out + (size_t)row0 * OUT_F + col;
      #pragma unroll
      for (int v = 0; v < 8; ++v)
        o[(size_t)v * OUT_F] = acc[mt][nt][v] + bv;
    }
  }
}

// ---------------------------------------------------------------- launch
extern "C" void kernel_launch(void* const* d_in, const int* in_sizes, int n_in,
                              void* d_out, int out_size, void* d_ws, size_t ws_size,
                              hipStream_t stream)
{
  const float* x      = (const float*)d_in[0];
  const int*   codes  = (const int*)d_in[1];
  const float* basis  = (const float*)d_in[2];
  const float* scales = (const float*)d_in[3];
  const float* bias   = (const float*)d_in[4];
  float*       out    = (float*)d_out;

  _Float16* Wh = (_Float16*)d_ws;                                    // 32 MiB
  _Float16* Xh = (_Float16*)((char*)d_ws + (size_t)OUT_F * IN_F * 2); // 64 MiB

  decode_w_kernel<<<(OUT_F * (size_t)IN_F / 4) / 256, 256, 0, stream>>>(
      codes, basis, scales, Wh);
  convert_x_kernel<<<((size_t)NROWS * IN_F / 8) / 256, 256, 0, stream>>>(x, Xh);

  dim3 grid(OUT_F / BN, NROWS / BM);   // 32 x 64 workgroups
  gemm_kernel<<<grid, 256, 0, stream>>>(Xh, Wh, bias, out);
}
